// EncoderLayer_56049323213809
// MI455X (gfx1250) — compile-verified, matrix operand order checked by reading, not run on hardware
//
#include <hip/hip_runtime.h>

// ---------------------------------------------------------------------------
// MI455X (gfx1250) transformer encoder layer.
// S=1024 B=4 D=1024 H=16 dh=64 DFF=4096.
// All heavy GEMMs use v_wmma_f32_16x16x32_bf16 (fp32 accumulate).
// Attention is fully fused (softmax is over the BATCH axis -> per-element),
// so the 256MB score tensor never reaches HBM.
// GEMM tile staging uses GLOBAL_LOAD_ASYNC_TO_LDS_B128 (ASYNCcnt) when the
// toolchain exposes the builtin; falls back to b128 load/ds_store otherwise.
// ---------------------------------------------------------------------------

typedef __attribute__((ext_vector_type(16))) __bf16 v16bf;
typedef __attribute__((ext_vector_type(8)))  float  v8f;
typedef __attribute__((ext_vector_type(4)))  int    v4i;

#if defined(__has_builtin)
#if __has_builtin(__builtin_amdgcn_global_load_async_to_lds_b128)
#define HAVE_ASYNC_LDS 1
#endif
#endif

union Frag {
    unsigned u[8];
    v16bf    v;
};

__device__ __forceinline__ v8f vzero() {
    v8f z;
    for (int i = 0; i < 8; ++i) z[i] = 0.f;
    return z;
}

__device__ __forceinline__ v8f wmma_bf16(const Frag& a, const Frag& b, v8f c) {
    return __builtin_amdgcn_wmma_f32_16x16x32_bf16(
        /*neg_a=*/false, a.v, /*neg_b=*/false, b.v,
        /*c_mod=*/(short)0, c, /*reuse_a=*/false, /*reuse_b=*/false);
}

// Per-lane K start (even) for VGPR v of a 16x32 bf16 A-operand (B mirrors with
// lane -> column).  From CDNA5 ISA 7.12.2 "16-bit A-Matrix 16x32".
__device__ __forceinline__ int frag_k0(int lane, int v) {
    int base = (lane < 16) ? 0 : 8;
    return (v < 4) ? (base + 2 * v) : (base + 16 + 2 * (v - 4));
}

__device__ __forceinline__ unsigned short f2bf(float f) {
    unsigned u = __builtin_bit_cast(unsigned, f);
    unsigned r = u + 0x7FFFu + ((u >> 16) & 1u);   // RNE
    return (unsigned short)(r >> 16);
}

__device__ __forceinline__ void wait_async0() {
#if defined(HAVE_ASYNC_LDS)
#if __has_builtin(__builtin_amdgcn_s_wait_asynccnt)
    __builtin_amdgcn_s_wait_asynccnt(0);
#else
    asm volatile("s_wait_asynccnt 0" ::: "memory");
#endif
#endif
}

// ---------------------------------------------------------------------------
// elementwise convert f32 -> bf16
// ---------------------------------------------------------------------------
__global__ __launch_bounds__(256) void cvt_bf16_kernel(const float* __restrict__ src,
                                                       unsigned short* __restrict__ dst,
                                                       int n) {
    int i = blockIdx.x * 256 + threadIdx.x;
    if (i < n) dst[i] = f2bf(src[i]);
}

// rel_w (B,H,dh,S) -> rwT[bh][d][c] = rel_w[b,h,c,S-1-d]  (bf16, c contiguous)
__global__ __launch_bounds__(256) void build_rwT_kernel(const float* __restrict__ rel_w,
                                                        unsigned short* __restrict__ rwT) {
    int i = blockIdx.x * 256 + threadIdx.x;            // i = ((bh*1024)+d)*64 + c
    int c  = i & 63;
    int d  = (i >> 6) & 1023;
    int bh = i >> 16;
    rwT[i] = f2bf(rel_w[((size_t)(bh * 64 + c)) * 1024 + (1023 - d)]);
}

// v (flat y layout) -> vT[bh][c][j] = v[b*2^20 + j*1024 + h*64 + c]  (bf16, j contig)
__global__ __launch_bounds__(256) void build_vT_kernel(const unsigned short* __restrict__ vb,
                                                       unsigned short* __restrict__ vT) {
    int i = blockIdx.x * 256 + threadIdx.x;            // i = ((bh*64)+c)*1024 + j
    int j  = i & 1023;
    int c  = (i >> 10) & 63;
    int bh = i >> 16;
    int b = bh >> 4, h = bh & 15;
    vT[i] = vb[(size_t)b * 1048576 + (size_t)j * 1024 + h * 64 + c];
}

// ---------------------------------------------------------------------------
// Generic bf16 WMMA GEMM:  C(MxN) = A(MxK) * Bt(NxK)^T  (both K-contiguous)
// BM=BN=128, BK=64, 256 threads = 8 waves (4 m-subwaves x 2 n-subwaves),
// wave tile 32x64 -> 8 f32 accumulator tiles per wave.
// Tile staging: async global->LDS (no VGPR round-trip) when available.
// ---------------------------------------------------------------------------
#define GBM 128
#define GBN 128
#define GBK 64
#define LSTR 80   // LDS row stride in bf16 (160B, 16B aligned, conflict-padded)

__device__ __forceinline__ void stage_tile_b128(const unsigned short* __restrict__ gsrc,
                                                unsigned short* __restrict__ ldst) {
#if defined(HAVE_ASYNC_LDS)
    __builtin_amdgcn_global_load_async_to_lds_b128(
        (__attribute__((address_space(1))) v4i*)(const void*)gsrc,
        (__attribute__((address_space(3))) v4i*)(void*)ldst,
        0, 0);
#else
    *(uint4*)ldst = *(const uint4*)gsrc;
#endif
}

template <bool OBF, bool OF32, bool BIAS, bool RELU, bool RESID>
__global__ __launch_bounds__(256) void gemm_bf16_kernel(
    const unsigned short* __restrict__ A,
    const unsigned short* __restrict__ Bt,
    float* __restrict__ Cf, unsigned short* __restrict__ Cb,
    const float* __restrict__ bias, const float* __restrict__ resid,
    int M, int N, int K) {
    __shared__ unsigned short Al[GBM * LSTR];
    __shared__ unsigned short Bl[GBN * LSTR];

    const int tid  = threadIdx.x;
    const int lane = tid & 31;
    const int w    = tid >> 5;
    const int wm   = w & 3;        // 4 waves along M (32 rows each)
    const int wn   = w >> 2;       // 2 waves along N (64 cols each)
    const int m0   = blockIdx.y * GBM;
    const int n0   = blockIdx.x * GBN;
    const int rl   = lane & 15;
    const int hi8  = (lane >> 4) << 3;

    v8f acc[2][4];
    for (int s = 0; s < 2; ++s)
        for (int t = 0; t < 4; ++t) acc[s][t] = vzero();

    const unsigned* AlU = (const unsigned*)Al;
    const unsigned* BlU = (const unsigned*)Bl;

    for (int k0 = 0; k0 < K; k0 += GBK) {
        // cooperative staging: 1024 chunks of 8 bf16 per tile, 4 per thread
        for (int c = tid; c < (GBM * GBK / 8); c += 256) {
            int row  = c >> 3;
            int col8 = (c & 7) << 3;
            stage_tile_b128(&A[(size_t)(m0 + row) * K + k0 + col8],
                            &Al[row * LSTR + col8]);
            stage_tile_b128(&Bt[(size_t)(n0 + row) * K + k0 + col8],
                            &Bl[row * LSTR + col8]);
        }
        wait_async0();
        __syncthreads();

        for (int ks = 0; ks < GBK; ks += 32) {
            Frag af[2], bf[4];
            for (int s = 0; s < 2; ++s) {
                int row = wm * 32 + s * 16 + rl;
                for (int v = 0; v < 8; ++v) {
                    int kk = ks + frag_k0(lane, v);
                    af[s].u[v] = AlU[row * (LSTR / 2) + (kk >> 1)];
                }
            }
            for (int t = 0; t < 4; ++t) {
                int col = wn * 64 + t * 16 + rl;
                for (int v = 0; v < 8; ++v) {
                    int kk = ks + frag_k0(lane, v);
                    bf[t].u[v] = BlU[col * (LSTR / 2) + (kk >> 1)];
                }
            }
            for (int s = 0; s < 2; ++s)
                for (int t = 0; t < 4; ++t)
                    acc[s][t] = wmma_bf16(af[s], bf[t], acc[s][t]);
        }
        __syncthreads();
    }

    // epilogue
    for (int s = 0; s < 2; ++s)
        for (int t = 0; t < 4; ++t) {
            int col = n0 + wn * 64 + t * 16 + rl;
            float bv = BIAS ? bias[col] : 0.f;
            for (int r = 0; r < 8; ++r) {
                int row = m0 + wm * 32 + s * 16 + r + hi8;
                float v = acc[s][t][r] + bv;
                if (RELU)  v = fmaxf(v, 0.f);
                if (RESID) v += resid[(size_t)row * N + col];
                if (OF32)  Cf[(size_t)row * N + col] = v;
                if (OBF)   Cb[(size_t)row * N + col] = f2bf(v);
            }
        }
}

// ---------------------------------------------------------------------------
// Fused attention.
// grid = (H=16, S/128=8); block = 256 (8 waves).  Wave w owns i-tile
// ib = blockIdx.y*128 + w*16 (16 rows) for ALL 4 batches.
// Per 32-wide j step and batch b:
//   score tile = 0.125 * q k^T  +  G[i, i-j]   (G = q @ col-reversed rel_w,
//   realized as two aligned 16-wide d-tiles + LDS diagonal gather)
//   softmax across b in registers, attn -> LDS (bf16) -> A-operand,
//   out[b] += attn @ v  via vT (K-contiguous).
// ---------------------------------------------------------------------------
__global__ __launch_bounds__(256) void attn_kernel(
    const unsigned short* __restrict__ qb,
    const unsigned short* __restrict__ kb,
    const unsigned short* __restrict__ rwT,
    const unsigned short* __restrict__ vT,
    float* __restrict__ attn_out) {
    __shared__ float          glds[8][16][32];
    __shared__ unsigned short alds[8][4][16][32];

    const int tid  = threadIdx.x;
    const int lane = tid & 31;
    const int w    = tid >> 5;
    const int h    = blockIdx.x;
    const int ib   = blockIdx.y * 128 + w * 16;
    const int rl   = lane & 15;
    const int hi8  = (lane >> 4) << 3;

    // q fragments for all 4 batches, both 32-wide K halves of dh=64
    unsigned qf[4][2][8];
    for (int b = 0; b < 4; ++b)
        for (int kk = 0; kk < 2; ++kk)
            for (int v = 0; v < 8; ++v) {
                int k0 = kk * 32 + frag_k0(lane, v);
                qf[b][kk][v] = *(const unsigned*)&qb[(size_t)b * 1048576 +
                                                    (size_t)(ib + rl) * 1024 + h * 64 + k0];
            }

    v8f oacc[4][4];
    for (int b = 0; b < 4; ++b)
        for (int t = 0; t < 4; ++t) oacc[b][t] = vzero();

    for (int j0 = 0; j0 < 1024; j0 += 32) {
        for (int half = 0; half < 2; ++half) {
            const int jt = j0 + half * 16;
            v8f sc[4];

            for (int b = 0; b < 4; ++b) {
                // --- q k^T tile ---
                v8f s = vzero();
                for (int kk = 0; kk < 2; ++kk) {
                    Frag kf, aq;
                    for (int v = 0; v < 8; ++v) {
                        int k0 = kk * 32 + frag_k0(lane, v);
                        kf.u[v] = *(const unsigned*)&kb[(size_t)b * 1048576 +
                                                        (size_t)(jt + rl) * 1024 + h * 64 + k0];
                        aq.u[v] = qf[b][kk][v];
                    }
                    s = wmma_bf16(aq, kf, s);
                }

                // --- two G d-tiles covering d = i-j for this (i-tile, j-tile) ---
                const int dt0 = ib - jt - 16;
                for (int g = 0; g < 2; ++g) {
                    int dt = dt0 + g * 16;
                    v8f ga = vzero();
                    if (dt + 15 >= 0) {
                        int d = dt + rl;  // this lane's G column
                        for (int kk = 0; kk < 2; ++kk) {
                            Frag rf, aq;
                            for (int v = 0; v < 8; ++v) {
                                int k0 = kk * 32 + frag_k0(lane, v);
                                rf.u[v] = (d >= 0)
                                    ? *(const unsigned*)&rwT[(size_t)((b * 16 + h) * 1024 + d) * 64 + k0]
                                    : 0u;
                                aq.u[v] = qf[b][kk][v];
                            }
                            ga = wmma_bf16(aq, rf, ga);
                        }
                    }
                    for (int r = 0; r < 8; ++r)
                        glds[w][r + hi8][g * 16 + rl] = ga[r];
                }
                asm volatile("s_wait_dscnt 0" ::: "memory");

                // diagonal gather: score += G[i, i-j] (0 when j > i)
                for (int r = 0; r < 8; ++r) {
                    int   rrel = r + hi8;
                    int   d    = ib + rrel - (jt + rl);
                    float gv   = (d >= 0) ? glds[w][rrel][rrel - rl + 16] : 0.f;
                    sc[b][r]   = s[r] * 0.125f + gv;
                }
            }

            // --- softmax over the batch axis (4 values per element) ---
            for (int r = 0; r < 8; ++r) {
                float m = sc[0][r];
                for (int b = 1; b < 4; ++b) m = fmaxf(m, sc[b][r]);
                float sum = 0.f;
                float e[4];
                for (int b = 0; b < 4; ++b) { e[b] = __expf(sc[b][r] - m); sum += e[b]; }
                float inv = 1.f / sum;
                for (int b = 0; b < 4; ++b)
                    alds[w][b][r + hi8][half * 16 + rl] = f2bf(e[b] * inv);
            }
        }
        asm volatile("s_wait_dscnt 0" ::: "memory");

        // --- out[b] += attn(16x32) @ v(32x64) ---
        for (int b = 0; b < 4; ++b) {
            Frag af;
            const unsigned* aU = (const unsigned*)alds[w][b];
            for (int v = 0; v < 8; ++v) {
                int k0 = frag_k0(lane, v);
                af.u[v] = aU[rl * 16 + (k0 >> 1)];
            }
            for (int t = 0; t < 4; ++t) {
                Frag vf;
                int c = t * 16 + rl;
                for (int v = 0; v < 8; ++v) {
                    int jk = j0 + frag_k0(lane, v);
                    vf.u[v] = *(const unsigned*)&vT[(size_t)((b * 16 + h) * 64 + c) * 1024 + jk];
                }
                oacc[b][t] = wmma_bf16(af, vf, oacc[b][t]);
            }
        }
    }

    // store out in (B,H,S,dh) flat order (== raw reshape to (S,B,D))
    for (int b = 0; b < 4; ++b)
        for (int t = 0; t < 4; ++t)
            for (int r = 0; r < 8; ++r) {
                int row = ib + r + hi8;
                attn_out[(size_t)b * 1048576 + (size_t)h * 65536 +
                         (size_t)row * 64 + t * 16 + rl] = oacc[b][t][r];
            }
}

// ---------------------------------------------------------------------------
// LayerNorm over D=1024: dst = LN(a (+ b)) * g + beta; optional bf16 copy.
// One block (256 threads, 8 waves) per row.
// ---------------------------------------------------------------------------
__global__ __launch_bounds__(256) void ln_kernel(
    const float* __restrict__ a, const float* __restrict__ badd,
    const float* __restrict__ g, const float* __restrict__ beta,
    float* __restrict__ dst, unsigned short* __restrict__ dstb) {
    __shared__ float partial[17];
    const int row  = blockIdx.x;
    const int tid  = threadIdx.x;
    const int lane = tid & 31;
    const int w    = tid >> 5;
    const float* pa = a + (size_t)row * 1024;
    const float* pb = badd ? badd + (size_t)row * 1024 : nullptr;

    float x[4], s = 0.f, s2 = 0.f;
    for (int i = 0; i < 4; ++i) {
        int d   = tid + i * 256;
        float v = pa[d];
        if (pb) v += pb[d];
        x[i] = v; s += v; s2 += v * v;
    }
    for (int m = 16; m > 0; m >>= 1) {
        s  += __shfl_xor(s, m, 32);
        s2 += __shfl_xor(s2, m, 32);
    }
    if (lane == 0) { partial[w] = s; partial[8 + w] = s2; }
    __syncthreads();
    if (tid == 0) {
        float ts = 0.f, t2 = 0.f;
        for (int i = 0; i < 8; ++i) { ts += partial[i]; t2 += partial[8 + i]; }
        partial[0]  = ts * (1.f / 1024.f);
        partial[16] = t2 * (1.f / 1024.f);
    }
    __syncthreads();
    float mean = partial[0];
    float var  = partial[16] - mean * mean;
    float inv  = rsqrtf(var + 1e-5f);
    for (int i = 0; i < 4; ++i) {
        int d   = tid + i * 256;
        float y = (x[i] - mean) * inv * g[d] + beta[d];
        dst[(size_t)row * 1024 + d] = y;
        if (dstb) dstb[(size_t)row * 1024 + d] = f2bf(y);
    }
}

// ---------------------------------------------------------------------------
extern "C" void kernel_launch(void* const* d_in, const int* in_sizes, int n_in,
                              void* d_out, int out_size, void* d_ws, size_t ws_size,
                              hipStream_t stream) {
    (void)in_sizes; (void)n_in; (void)out_size; (void)ws_size;

    const float* x     = (const float*)d_in[0];
    const float* wq    = (const float*)d_in[1];
    const float* wk    = (const float*)d_in[2];
    const float* wv    = (const float*)d_in[3];
    const float* rel_w = (const float*)d_in[4];
    const float* ln1_g = (const float*)d_in[5];
    const float* ln1_b = (const float*)d_in[6];
    const float* ln2_g = (const float*)d_in[7];
    const float* ln2_b = (const float*)d_in[8];
    const float* w1_w  = (const float*)d_in[9];
    const float* w1_b  = (const float*)d_in[10];
    const float* w2_w  = (const float*)d_in[11];
    const float* w2_b  = (const float*)d_in[12];
    float* out = (float*)d_out;

    const size_t M4 = 1048576 * 4;   // 4M elements
    const size_t M1 = 1048576;       // 1M elements
    const size_t M16 = 1048576 * 16; // 16M elements

    size_t off = 0;
    auto carve = [&](size_t bytes) -> char* {
        char* p = (char*)d_ws + off;
        off += (bytes + 255) & ~(size_t)255;
        return p;
    };
    unsigned short* xb   = (unsigned short*)carve(M4 * 2);
    unsigned short* wqb  = (unsigned short*)carve(M1 * 2);
    unsigned short* wkb  = (unsigned short*)carve(M1 * 2);
    unsigned short* wvb  = (unsigned short*)carve(M1 * 2);
    unsigned short* w1b  = (unsigned short*)carve(M4 * 2);
    unsigned short* w2b  = (unsigned short*)carve(M4 * 2);
    unsigned short* rwT  = (unsigned short*)carve(M4 * 2);
    unsigned short* qb   = (unsigned short*)carve(M4 * 2);
    unsigned short* kb   = (unsigned short*)carve(M4 * 2);
    unsigned short* vb   = (unsigned short*)carve(M4 * 2);
    unsigned short* vTb  = (unsigned short*)carve(M4 * 2);
    float*          aout = (float*)carve(M4 * 4);
    float*          h1   = (float*)carve(M4 * 4);
    unsigned short* h1b  = (unsigned short*)carve(M4 * 2);
    unsigned short* f1b  = (unsigned short*)carve(M16 * 2);
    float*          h2   = (float*)carve(M4 * 4);

    // 1) bf16 conversions
    cvt_bf16_kernel<<<(int)(M4 / 256), 256, 0, stream>>>(x, xb, (int)M4);
    cvt_bf16_kernel<<<(int)(M1 / 256), 256, 0, stream>>>(wq, wqb, (int)M1);
    cvt_bf16_kernel<<<(int)(M1 / 256), 256, 0, stream>>>(wk, wkb, (int)M1);
    cvt_bf16_kernel<<<(int)(M1 / 256), 256, 0, stream>>>(wv, wvb, (int)M1);
    cvt_bf16_kernel<<<(int)(M4 / 256), 256, 0, stream>>>(w1_w, w1b, (int)M4);
    cvt_bf16_kernel<<<(int)(M4 / 256), 256, 0, stream>>>(w2_w, w2b, (int)M4);
    build_rwT_kernel<<<(int)(M4 / 256), 256, 0, stream>>>(rel_w, rwT);

    // 2) Q,K,V projections (M=4096, N=1024, K=1024), bf16 outputs
    dim3 gqkv(1024 / GBN, 4096 / GBM);
    gemm_bf16_kernel<true, false, false, false, false><<<gqkv, 256, 0, stream>>>(
        xb, wqb, nullptr, qb, nullptr, nullptr, 4096, 1024, 1024);
    gemm_bf16_kernel<true, false, false, false, false><<<gqkv, 256, 0, stream>>>(
        xb, wkb, nullptr, kb, nullptr, nullptr, 4096, 1024, 1024);
    gemm_bf16_kernel<true, false, false, false, false><<<gqkv, 256, 0, stream>>>(
        xb, wvb, nullptr, vb, nullptr, nullptr, 4096, 1024, 1024);
    build_vT_kernel<<<(int)(M4 / 256), 256, 0, stream>>>(vb, vTb);

    // 3) fused attention (scores + skew + batch-softmax + attn@v)
    attn_kernel<<<dim3(16, 8), 256, 0, stream>>>(qb, kb, rwT, vTb, aout);

    // 4) h1 = LN(x + attn_out)  (+ bf16 copy for FFN)
    ln_kernel<<<4096, 256, 0, stream>>>(x, aout, ln1_g, ln1_b, h1, h1b);

    // 5) FFN1: relu(h1 @ w1^T + b1) -> bf16   (M=4096, N=4096, K=1024)
    gemm_bf16_kernel<true, false, true, true, false><<<dim3(4096 / GBN, 4096 / GBM), 256, 0, stream>>>(
        h1b, w1b, nullptr, f1b, w1_b, nullptr, 4096, 4096, 1024);

    // 6) FFN2: h2 = h1 + (f1 @ w2^T + b2)     (M=4096, N=1024, K=4096)
    gemm_bf16_kernel<false, true, true, false, true><<<dim3(1024 / GBN, 4096 / GBM), 256, 0, stream>>>(
        f1b, w2b, h2, nullptr, w2_b, h1, 4096, 1024, 4096);

    // 7) out = LN(h2)
    ln_kernel<<<4096, 256, 0, stream>>>(h2, nullptr, ln2_g, ln2_b, out, nullptr);
}
